// CAM_Module_3831110828713
// MI455X (gfx1250) — matrix operand order repaired and checked by reading, not compiled, not run on hardware
//
#include <hip/hip_runtime.h>
#include <hip/hip_bf16.h>

// CAM module for MI455X (gfx1250): bf16 WMMA path, double-buffered LDS pipeline.
//   energy  = X X^T   (symmetric: only 10 of 16 tiles computed, mirrored)
//   att     = softmax(rowmax(energy) - energy) = exp(rowmin - e)/sum  -> bf16
//   out     = gamma * (att X) + x   (fused epilogue)

typedef __attribute__((ext_vector_type(16))) __bf16 v16bf;
typedef __attribute__((ext_vector_type(8)))  float  v8f;

#define B_DIM 32
#define C_DIM 512
#define L_DIM 4096
#define TILE  128      // output tile per workgroup (M and N)
#define KS    32       // K-slab per iteration (one bf16 WMMA deep)
#define PADK  40       // padded LDS row stride in bf16 elems (80 B, 16B aligned)

// upper-triangular tile map for the symmetric energy GEMM (4x4 -> 10 tiles)
__device__ const int cTM[10] = {0,0,0,0,1,1,1,2,2,3};
__device__ const int cTN[10] = {0,1,2,3,1,2,3,2,3,3};

union Frag { v16bf v; uint4 q[2]; };

// A fragment (16x32 bf16): lane&15 -> M; lanes<16 hold K=0..7,16..23;
// lanes>=16 hold K=8..15,24..31 (ISA 7.12.2). LDS tile is [row][k].
__device__ __forceinline__ v16bf load_frag_a(const __bf16* t, int row, int lane) {
    const char* p = (const char*)(t + row * PADK);
    const int o = (lane < 16) ? 0 : 16;
    Frag f;
    f.q[0] = *(const uint4*)(p + o);
    f.q[1] = *(const uint4*)(p + o + 32);
    return f.v;
}

// B fragment (32x16 bf16): lane&15 -> N; lanes<16 hold K=0..7,16..23; lanes>=16 K=8..15,24..31.
__device__ __forceinline__ v16bf load_frag_b(const __bf16* t, int col, int lane) {
    const char* p = (const char*)(t + col * PADK);
    const int o = (lane < 16) ? 0 : 16;
    Frag f;
    f.q[0] = *(const uint4*)(p + o);
    f.q[1] = *(const uint4*)(p + o + 32);
    return f.v;
}

// f32x16 registers -> bf16 LDS row (compiler emits v_cvt_pk_bf16_f32 + ds_store_b128)
__device__ __forceinline__ void stage_store(__bf16* dst, const float4* r) {
    #pragma unroll
    for (int u = 0; u < 4; ++u) {
        float4 f = r[u];
        dst[u*4+0] = (__bf16)f.x; dst[u*4+1] = (__bf16)f.y;
        dst[u*4+2] = (__bf16)f.z; dst[u*4+3] = (__bf16)f.w;
    }
}

// ---------------------------------------------------------------------------
// Kernel 1: energy[b] = X_b * X_b^T  (C x C, K = L), upper-triangular tiles
// grid: (10, 1, B), block: 256 (8 waves, 2x4 -> each wave 64x32)
// ---------------------------------------------------------------------------
__global__ __launch_bounds__(256)
void cam_energy_kernel(const float* __restrict__ x, float* __restrict__ energy) {
    __shared__ __align__(16) __bf16 ldsA[2][TILE * PADK];
    __shared__ __align__(16) __bf16 ldsB[2][TILE * PADK];

    const int b    = blockIdx.z;
    const int m0   = cTM[blockIdx.x] * TILE;
    const int n0   = cTN[blockIdx.x] * TILE;
    const int tid  = threadIdx.x;
    const int lane = tid & 31;
    const int wave = tid >> 5;
    const int wm   = (wave >> 2) * 64;
    const int wn   = (wave & 3) * 32;

    const float* xb = x + (size_t)b * C_DIM * L_DIM;

    v8f acc[4][2] = {};

    // staging: thread t covers row (t>>1), 16 floats at col (t&1)*16
    const int lr = tid >> 1;
    const int lc = (tid & 1) * 16;
    const float* pa = xb + (size_t)(m0 + lr) * L_DIM + lc;
    const float* pb = xb + (size_t)(n0 + lr) * L_DIM + lc;
    __bf16* const sa = &ldsA[0][0] + lr * PADK + lc;
    __bf16* const sb = &ldsB[0][0] + lr * PADK + lc;

    float4 ra[4], rb[4];
    #pragma unroll
    for (int u = 0; u < 4; ++u) {
        ra[u] = ((const float4*)pa)[u];
        rb[u] = ((const float4*)pb)[u];
    }
    stage_store(sa, ra);
    stage_store(sb, rb);

    const int NIT = L_DIM / KS;
    for (int it = 0; it < NIT; ++it) {
        const int cur = it & 1;
        const bool more = (it + 1) < NIT;
        if (more) {   // prefetch next slab into registers (overlaps compute)
            const float* qa = pa + (size_t)(it + 1) * KS;
            const float* qb = pb + (size_t)(it + 1) * KS;
            #pragma unroll
            for (int u = 0; u < 4; ++u) {
                ra[u] = ((const float4*)qa)[u];
                rb[u] = ((const float4*)qb)[u];
            }
        }
        __syncthreads();   // buf[cur] stores done; prev readers of buf[cur^1] done

        v16bf bf[2];
        #pragma unroll
        for (int j = 0; j < 2; ++j)
            bf[j] = load_frag_b(ldsB[cur], wn + j*16 + (lane & 15), lane);
        #pragma unroll
        for (int i = 0; i < 4; ++i) {
            v16bf af = load_frag_a(ldsA[cur], wm + i*16 + (lane & 15), lane);
            #pragma unroll
            for (int j = 0; j < 2; ++j)
                acc[i][j] = __builtin_amdgcn_wmma_f32_16x16x32_bf16(
                    false, af, false, bf[j], (short)0, acc[i][j], false, false);
        }

        if (more) {   // drain staged registers into the other buffer
            stage_store(sa + (cur ^ 1) * (TILE * PADK), ra);
            stage_store(sb + (cur ^ 1) * (TILE * PADK), rb);
        }
    }

    // C/D layout: VGPR e, lanes<16 -> M=e, lanes>=16 -> M=e+8; N = lane&15
    float* eb = energy + (size_t)b * C_DIM * C_DIM;
    const bool mirror = (m0 != n0);
    #pragma unroll
    for (int i = 0; i < 4; ++i) {
        const int mb = m0 + wm + i*16 + ((lane < 16) ? 0 : 8);
        #pragma unroll
        for (int j = 0; j < 2; ++j) {
            const int col = n0 + wn + j*16 + (lane & 15);
            #pragma unroll
            for (int e = 0; e < 8; ++e) {
                const float v = acc[i][j][e];
                eb[(size_t)(mb + e) * C_DIM + col] = v;
                if (mirror)
                    eb[(size_t)col * C_DIM + (mb + e)] = v;   // symmetric mirror
            }
        }
    }
}

// ---------------------------------------------------------------------------
// Kernel 2: attention = softmax(rowmax - energy) = exp(rowmin - e) / sum(...)
// one 256-thread block per row of 512
// ---------------------------------------------------------------------------
__global__ __launch_bounds__(256)
void cam_softmax_kernel(const float* __restrict__ energy, __bf16* __restrict__ att) {
    __shared__ float red[256];
    const size_t row = blockIdx.x;
    const float* e = energy + row * C_DIM;
    const int tid = threadIdx.x;

    const float v0 = e[tid];
    const float v1 = e[tid + 256];

    red[tid] = fminf(v0, v1);
    __syncthreads();
    #pragma unroll
    for (int s = 128; s > 0; s >>= 1) {
        if (tid < s) red[tid] = fminf(red[tid], red[tid + s]);
        __syncthreads();
    }
    const float mn = red[0];
    __syncthreads();

    const float p0 = __expf(mn - v0);
    const float p1 = __expf(mn - v1);
    red[tid] = p0 + p1;
    __syncthreads();
    #pragma unroll
    for (int s = 128; s > 0; s >>= 1) {
        if (tid < s) red[tid] += red[tid + s];
        __syncthreads();
    }
    const float inv = 1.0f / red[0];

    __bf16* a = att + row * C_DIM;
    a[tid]       = (__bf16)(p0 * inv);
    a[tid + 256] = (__bf16)(p1 * inv);
}

// ---------------------------------------------------------------------------
// Kernel 3: out[b] = gamma * (att_b @ X_b) + X_b   (C x L, K = C)
// grid: (L/TILE, C/TILE, B), block: 256, double-buffered pipeline
// ---------------------------------------------------------------------------
__global__ __launch_bounds__(256)
void cam_out_kernel(const float* __restrict__ x, const __bf16* __restrict__ att,
                    const float* __restrict__ gamma, float* __restrict__ out) {
    __shared__ __align__(16) __bf16 ldsA[2][TILE * PADK];  // att tile [m][k]
    __shared__ __align__(16) __bf16 ldsB[2][TILE * PADK];  // x tile transposed [n][k]

    const int b    = blockIdx.z;
    const int m0   = blockIdx.y * TILE;     // over C
    const int n0   = blockIdx.x * TILE;     // over L
    const int tid  = threadIdx.x;
    const int lane = tid & 31;
    const int wave = tid >> 5;
    const int wm   = (wave >> 2) * 64;
    const int wn   = (wave & 3) * 32;

    const __bf16* ab = att + (size_t)b * C_DIM * C_DIM;
    const float*  xb = x   + (size_t)b * C_DIM * L_DIM;

    v8f acc[4][2] = {};

    const int lr = tid >> 1;          // A staging: row, 16 bf16 at (t&1)*16
    const int lc = (tid & 1) * 16;
    const int bk = tid >> 3;          // B staging: k within slab (0..31)
    const int bn = (tid & 7) * 16;    // B staging: n sub-block of 16

    const __bf16* pa = ab + (size_t)(m0 + lr) * C_DIM + lc;
    const float*  pb = xb + (size_t)bk * L_DIM + n0 + bn;
    __bf16* const sa = &ldsA[0][0] + lr * PADK + lc;

    uint4  ra[2];
    float4 rb[4];
    ra[0] = ((const uint4*)pa)[0];
    ra[1] = ((const uint4*)pa)[1];
    #pragma unroll
    for (int u = 0; u < 4; ++u) rb[u] = ((const float4*)pb)[u];

    // initial stage -> buffer 0
    ((uint4*)sa)[0] = ra[0];
    ((uint4*)sa)[1] = ra[1];
    #pragma unroll
    for (int u = 0; u < 4; ++u) {
        float4 f = rb[u];
        ldsB[0][(bn + u*4 + 0) * PADK + bk] = (__bf16)f.x;
        ldsB[0][(bn + u*4 + 1) * PADK + bk] = (__bf16)f.y;
        ldsB[0][(bn + u*4 + 2) * PADK + bk] = (__bf16)f.z;
        ldsB[0][(bn + u*4 + 3) * PADK + bk] = (__bf16)f.w;
    }

    const int NIT = C_DIM / KS;   // 16
    for (int it = 0; it < NIT; ++it) {
        const int cur = it & 1;
        const bool more = (it + 1) < NIT;
        if (more) {   // prefetch next slab into registers
            const __bf16* qa = pa + (size_t)(it + 1) * KS;
            const float*  qb = pb + (size_t)(it + 1) * KS * L_DIM;
            ra[0] = ((const uint4*)qa)[0];
            ra[1] = ((const uint4*)qa)[1];
            #pragma unroll
            for (int u = 0; u < 4; ++u) rb[u] = ((const float4*)qb)[u];
        }
        __syncthreads();

        v16bf bf[2];
        #pragma unroll
        for (int j = 0; j < 2; ++j)
            bf[j] = load_frag_b(ldsB[cur], wn + j*16 + (lane & 15), lane);
        #pragma unroll
        for (int i = 0; i < 4; ++i) {
            v16bf af = load_frag_a(ldsA[cur], wm + i*16 + (lane & 15), lane);
            #pragma unroll
            for (int j = 0; j < 2; ++j)
                acc[i][j] = __builtin_amdgcn_wmma_f32_16x16x32_bf16(
                    false, af, false, bf[j], (short)0, acc[i][j], false, false);
        }

        if (more) {   // drain into the other buffer
            const int nxt = cur ^ 1;
            uint4* da = (uint4*)(sa + nxt * (TILE * PADK));
            da[0] = ra[0];
            da[1] = ra[1];
            #pragma unroll
            for (int u = 0; u < 4; ++u) {
                float4 f = rb[u];
                ldsB[nxt][(bn + u*4 + 0) * PADK + bk] = (__bf16)f.x;
                ldsB[nxt][(bn + u*4 + 1) * PADK + bk] = (__bf16)f.y;
                ldsB[nxt][(bn + u*4 + 2) * PADK + bk] = (__bf16)f.z;
                ldsB[nxt][(bn + u*4 + 3) * PADK + bk] = (__bf16)f.w;
            }
        }
    }

    const float g = gamma[0];
    float* ob = out + (size_t)b * C_DIM * L_DIM;
    #pragma unroll
    for (int i = 0; i < 4; ++i) {
        const int mb = m0 + wm + i*16 + ((lane < 16) ? 0 : 8);
        #pragma unroll
        for (int j = 0; j < 2; ++j) {
            const int col = n0 + wn + j*16 + (lane & 15);
            #pragma unroll
            for (int e = 0; e < 8; ++e) {
                const size_t idx = (size_t)(mb + e) * L_DIM + col;
                ob[idx] = g * acc[i][j][e] + xb[idx];
            }
        }
    }
}

// ---------------------------------------------------------------------------
extern "C" void kernel_launch(void* const* d_in, const int* in_sizes, int n_in,
                              void* d_out, int out_size, void* d_ws, size_t ws_size,
                              hipStream_t stream) {
    const float* x     = (const float*)d_in[0];
    const float* gamma = (const float*)d_in[1];
    float* out = (float*)d_out;

    // workspace: [energy f32: B*C*C] [attention bf16: B*C*C]
    float*  energy = (float*)d_ws;
    __bf16* att    = (__bf16*)((char*)d_ws + (size_t)B_DIM * C_DIM * C_DIM * sizeof(float));

    dim3 g1(10, 1, B_DIM);                        // upper-triangular 128x128 tiles
    cam_energy_kernel<<<g1, 256, 0, stream>>>(x, energy);

    dim3 g2(B_DIM * C_DIM);                       // 16384 rows
    cam_softmax_kernel<<<g2, 256, 0, stream>>>(energy, att);

    dim3 g3(L_DIM / TILE, C_DIM / TILE, B_DIM);   // 32 x 4 x 32
    cam_out_kernel<<<g3, 256, 0, stream>>>(x, att, gamma, out);
}
